// EncoderBlock_24678882083141
// MI455X (gfx1250) — compile-verified, matrix operand order checked by reading, not run on hardware
//
#include <hip/hip_runtime.h>
#include <hip/hip_bf16.h>

// Transformer encoder block for MI455X (gfx1250, wave32, WMMA + TDM).
// - all GEMMs: v_wmma_f32_16x16x32_bf16, fp32 accumulation
// - weights pre-transposed to [N,K] bf16 so A and B fragments are both
//   contiguous ds_load_b128 (no LDS scatter)
// - GEMM and flash attention are double-buffered: next tiles fed by the
//   Tensor Data Mover (tensor_load_to_lds / TENSORcnt) while WMMAs consume
//   the current buffer; one barrier per K-step / key-block.
// - flash attention: the 128MB [B,H,S,S] tensor never exists, so the whole
//   working set (~70MB bf16) stays inside the 192MB L2.

#define B_  2
#define S_  2048
#define D_  512
#define H_  8
#define DH_ 64
#define FF_ 2048
#define NT_ (B_ * S_)   // 4096 tokens

typedef __attribute__((ext_vector_type(16))) __bf16 bf16x16;
typedef __attribute__((ext_vector_type(8)))  __bf16 bf16x8;
typedef __attribute__((ext_vector_type(8)))  float  f32x8;
typedef __attribute__((ext_vector_type(4)))  unsigned int u32x4;
typedef __attribute__((ext_vector_type(8)))  int i32x8;
typedef __attribute__((ext_vector_type(4)))  int i32x4;

#if defined(__gfx1250__) && __has_builtin(__builtin_amdgcn_tensor_load_to_lds)
#define ATHENA_TDM 1
#else
#define ATHENA_TDM 0
#endif

__device__ __forceinline__ f32x8 wmma_bf16f32(bf16x16 a, bf16x16 b, f32x8 c) {
  // 8 args: (neg_a, A, neg_b, B, c_mod, C, reuse_a, reuse_b)
  return __builtin_amdgcn_wmma_f32_16x16x32_bf16(false, a, false, b, (short)0, c,
                                                 false, false);
}

// Build a 16-half A/B fragment from two contiguous 8-half (16B) chunks.
// CDNA5 wave32 16-bit operand layout: half h of the wave holds K=kb..kb+7 and
// K=kb+16..kb+23 with kb = 8*h.
__device__ __forceinline__ bf16x16 make_frag(const __bf16* p0, const __bf16* p1) {
  bf16x8 lo = *(const bf16x8*)p0;
  bf16x8 hi = *(const bf16x8*)p1;
  bf16x16 r;
#pragma unroll
  for (int i = 0; i < 8; ++i) { r[i] = lo[i]; r[i + 8] = hi[i]; }
  return r;
}

#if ATHENA_TDM
// 2D tile DMA: global (row-major, `stride_elems` 2-byte elems per row) -> LDS
// (packed tile_d0*tile_d1 elems). Builds a real D# (group0+group1), groups 2/3
// zero (2D tensor). Issued once per calling wave; completion via TENSORcnt.
__device__ __forceinline__ void tdm_load_2d(unsigned lds_byte_addr,
                                            const void* gaddr,
                                            unsigned tensor_d0, unsigned tensor_d1,
                                            unsigned tile_d0, unsigned tile_d1,
                                            unsigned stride_elems) {
  unsigned long long ga = (unsigned long long)(uintptr_t)gaddr;
  u32x4 g0;
  g0[0] = 1u;                                    // count=1, user descriptor
  g0[1] = lds_byte_addr;                         // LDS destination
  g0[2] = (unsigned)(ga & 0xffffffffu);          // global_addr[31:0]
  g0[3] = (unsigned)((ga >> 32) & 0x01ffffffu) | (2u << 30);  // addr[56:32]|type=2
  i32x8 g1;
  g1[0] = (int)(1u << 16);                       // data_size=1 (2 bytes)
  g1[1] = (int)((tensor_d0 & 0xffffu) << 16);    // tensor_dim0[15:0] @ bit48
  g1[2] = (int)((tensor_d0 >> 16) | ((tensor_d1 & 0xffffu) << 16));
  g1[3] = (int)((tensor_d1 >> 16) | (tile_d0 << 16));  // tile_dim0 @ bit112
  g1[4] = (int)(tile_d1 & 0xffffu);              // tile_dim1; tile_dim2=0
  g1[5] = (int)stride_elems;                     // tensor_dim0_stride[31:0]
  g1[6] = 0;
  g1[7] = 0;
  i32x4 z4 = {0, 0, 0, 0};
#if __clang_major__ >= 23
  i32x8 z8 = {0, 0, 0, 0, 0, 0, 0, 0};
  __builtin_amdgcn_tensor_load_to_lds(g0, g1, z4, z4, z8, 0);
#else
  __builtin_amdgcn_tensor_load_to_lds(g0, g1, z4, z4, 0);
#endif
}
#endif  // ATHENA_TDM

// ---------------------------------------------------------------------------
// GEMM: C[M,N] = A[M,K] * Bt[N,K]^T (+bias)(relu?)(+residual)
// Block 128 threads = 4 waves (2x2), block tile 64x128, K-step 32.
// Each wave: 32x64 patch = 2x4 WMMA tiles = 8 WMMA per K-step.
// ---------------------------------------------------------------------------
__global__ __launch_bounds__(128)
void gemm_bf16_kernel(const __bf16* __restrict__ A, const __bf16* __restrict__ Bt,
                      int M, int N, int K,
                      const float* __restrict__ bias,
                      const float* __restrict__ residual,
                      int do_relu,
                      float* __restrict__ outF,
                      __bf16* __restrict__ outB) {
  __shared__ __align__(16) __bf16 ldsA[2][64][32];    // [m][k]
  __shared__ __align__(16) __bf16 ldsB[2][128][32];   // [n][k]

  const int t    = threadIdx.x;
  const int lane = t & 31;
  const int wave = t >> 5;
  const int half = lane >> 4;
  const int r    = lane & 15;
  const int kb8  = half * 8;

  const int n0 = blockIdx.x * 128;
  const int m0 = blockIdx.y * 64;
  const int wm = (wave & 1) * 32;
  const int wn = (wave >> 1) * 64;

  f32x8 acc[2][4];
#pragma unroll
  for (int mi = 0; mi < 2; ++mi)
#pragma unroll
    for (int ni = 0; ni < 4; ++ni)
#pragma unroll
      for (int v = 0; v < 8; ++v) acc[mi][ni][v] = 0.0f;

  const __bf16* Abase = A + (size_t)m0 * K;
  const __bf16* Bbase = Bt + (size_t)n0 * K;

  // ---- prologue: fill buffer 0 with the k0=0 tiles --------------------------
#if ATHENA_TDM
  if (wave == 0) {
    tdm_load_2d((unsigned)(uintptr_t)&ldsA[0][0][0], Abase,
                (unsigned)K, (unsigned)M, 32u, 64u, (unsigned)K);
  } else if (wave == 1) {
    tdm_load_2d((unsigned)(uintptr_t)&ldsB[0][0][0], Bbase,
                (unsigned)K, (unsigned)N, 32u, 128u, (unsigned)K);
  }
  __builtin_amdgcn_s_wait_tensorcnt(0);
  __syncthreads();
#else
#pragma unroll
  for (int i = 0; i < 2; ++i) {
    int c = i * 128 + t;
    int row = c >> 2, ko = (c & 3) * 8;
    *(bf16x8*)&ldsA[0][row][ko] = *(const bf16x8*)(Abase + (size_t)row * K + ko);
  }
#pragma unroll
  for (int i = 0; i < 4; ++i) {
    int c = i * 128 + t;
    int row = c >> 2, ko = (c & 3) * 8;
    *(bf16x8*)&ldsB[0][row][ko] = *(const bf16x8*)(Bbase + (size_t)row * K + ko);
  }
  __syncthreads();
#endif

  int cur = 0;
  for (int k0 = 0; k0 < K; k0 += 32) {
    const int nb = cur ^ 1;
    const bool haveNext = (k0 + 32) < K;

    // ---- (a) start next tile fill (no waits yet) ----------------------------
#if ATHENA_TDM
    if (haveNext) {
      if (wave == 0) {
        tdm_load_2d((unsigned)(uintptr_t)&ldsA[nb][0][0], Abase + k0 + 32,
                    (unsigned)K, (unsigned)M, 32u, 64u, (unsigned)K);
      } else if (wave == 1) {
        tdm_load_2d((unsigned)(uintptr_t)&ldsB[nb][0][0], Bbase + k0 + 32,
                    (unsigned)K, (unsigned)N, 32u, 128u, (unsigned)K);
      }
    }
#else
    bf16x8 ra[2], rb[4];
    if (haveNext) {
#pragma unroll
      for (int i = 0; i < 2; ++i) {
        int c = i * 128 + t;
        int row = c >> 2, ko = (c & 3) * 8;
        ra[i] = *(const bf16x8*)(Abase + (size_t)row * K + k0 + 32 + ko);
      }
#pragma unroll
      for (int i = 0; i < 4; ++i) {
        int c = i * 128 + t;
        int row = c >> 2, ko = (c & 3) * 8;
        rb[i] = *(const bf16x8*)(Bbase + (size_t)row * K + k0 + 32 + ko);
      }
    }
#endif

    // ---- (b) math: 8 WMMAs against buffer `cur` -----------------------------
    bf16x16 afr[2], bfr[4];
#pragma unroll
    for (int mi = 0; mi < 2; ++mi) {
      const __bf16* pa = &ldsA[cur][wm + mi * 16 + r][0];
      afr[mi] = make_frag(pa + kb8, pa + kb8 + 16);
    }
#pragma unroll
    for (int ni = 0; ni < 4; ++ni) {
      const __bf16* pb = &ldsB[cur][wn + ni * 16 + r][0];
      bfr[ni] = make_frag(pb + kb8, pb + kb8 + 16);
    }
#pragma unroll
    for (int mi = 0; mi < 2; ++mi)
#pragma unroll
      for (int ni = 0; ni < 4; ++ni)
        acc[mi][ni] = wmma_bf16f32(afr[mi], bfr[ni], acc[mi][ni]);

    // ---- (c) finish next tile fill ------------------------------------------
#if ATHENA_TDM
    __builtin_amdgcn_s_wait_tensorcnt(0);
#else
    if (haveNext) {
#pragma unroll
      for (int i = 0; i < 2; ++i) {
        int c = i * 128 + t;
        int row = c >> 2, ko = (c & 3) * 8;
        *(bf16x8*)&ldsA[nb][row][ko] = ra[i];
      }
#pragma unroll
      for (int i = 0; i < 4; ++i) {
        int c = i * 128 + t;
        int row = c >> 2, ko = (c & 3) * 8;
        *(bf16x8*)&ldsB[nb][row][ko] = rb[i];
      }
    }
#endif
    __syncthreads();
    cur = nb;
  }

  // ---- epilogue. C layout: VGPR v <-> row v + 8*half, col = lane&15 ---------
#pragma unroll
  for (int mi = 0; mi < 2; ++mi) {
#pragma unroll
    for (int ni = 0; ni < 4; ++ni) {
      int colg = n0 + wn + ni * 16 + r;
      float bv = bias ? bias[colg] : 0.0f;
#pragma unroll
      for (int v = 0; v < 8; ++v) {
        int rowg = m0 + wm + mi * 16 + v + half * 8;
        float val = acc[mi][ni][v] + bv;
        if (do_relu) val = fmaxf(val, 0.0f);
        if (residual) val += residual[(size_t)rowg * N + colg];
        if (outF) outF[(size_t)rowg * N + colg] = val;
        if (outB) outB[(size_t)rowg * N + colg] = (__bf16)val;
      }
    }
  }
}

// ---------------------------------------------------------------------------
// Flash attention, double-buffered. Q/K/V/Y bf16 [B*S, H*DH] row-major.
// Grid (S/64, B*H), 128 threads (4 waves x 16 query rows).
// K tile fed by TDM (strided 2D descriptor); V tile register-staged and
// stored transposed (TDM cannot transpose). One barrier per key-block.
// ---------------------------------------------------------------------------
__global__ __launch_bounds__(128)
void flash_attn_kernel(const __bf16* __restrict__ Qg, const __bf16* __restrict__ Kg,
                       const __bf16* __restrict__ Vg, __bf16* __restrict__ Yg) {
  __shared__ __align__(16) __bf16 ldsK[2][64][64];    // [key][e]  B-frags QK^T
  __shared__ __align__(16) __bf16 ldsVt[2][64][64];   // [e][key]  B-frags PV
  __shared__ __align__(16) __bf16 ldsP[4][16][64];    // wave-private P staging

  const int t    = threadIdx.x;
  const int lane = t & 31;
  const int wave = t >> 5;
  const int half = lane >> 4;
  const int r    = lane & 15;
  const int kb8  = half * 8;

  const int b  = blockIdx.y >> 3;
  const int h  = blockIdx.y & 7;
  const int q0 = blockIdx.x * 64 + wave * 16;
  const int LD = H_ * DH_;

  const __bf16* qrow = Qg + (size_t)(b * S_ + q0 + r) * LD + h * DH_;
  bf16x16 aq[2];
#pragma unroll
  for (int kk = 0; kk < 2; ++kk)
    aq[kk] = make_frag(qrow + kk * 32 + kb8, qrow + kk * 32 + kb8 + 16);

  float m[8], l[8];
  f32x8 o[4];
#pragma unroll
  for (int v = 0; v < 8; ++v) { m[v] = -1e30f; l[v] = 0.0f; }
#pragma unroll
  for (int te = 0; te < 4; ++te)
#pragma unroll
    for (int v = 0; v < 8; ++v) o[te][v] = 0.0f;

  const int kr = t >> 4;         // 0..7
  const int eg = (t & 15) * 4;   // 0..60
  const __bf16* Kbase = Kg + (size_t)b * S_ * LD + h * DH_;
  const __bf16* Vbase = Vg + (size_t)b * S_ * LD + h * DH_;

  // ---- prologue: fill buffer 0 with key-block 0 -----------------------------
#if ATHENA_TDM
  if (wave == 0) {
    tdm_load_2d((unsigned)(uintptr_t)&ldsK[0][0][0], Kbase,
                (unsigned)LD, (unsigned)NT_, 64u, 64u, (unsigned)LD);
  }
#else
#pragma unroll
  for (int i = 0; i < 8; ++i) {
    int key = i * 8 + kr;
    *(uint2*)&ldsK[0][key][eg] = *(const uint2*)(Kbase + (size_t)key * LD + eg);
  }
#endif
#pragma unroll
  for (int i = 0; i < 8; ++i) {
    int key = i * 8 + kr;
    __bf16 tmp4[4];
    *(uint2*)tmp4 = *(const uint2*)(Vbase + (size_t)key * LD + eg);
#pragma unroll
    for (int j = 0; j < 4; ++j) ldsVt[0][eg + j][key] = tmp4[j];
  }
#if ATHENA_TDM
  __builtin_amdgcn_s_wait_tensorcnt(0);
#endif
  __syncthreads();

  int cur = 0;
  for (int key0 = 0; key0 < S_; key0 += 64) {
    const int nb = cur ^ 1;
    const bool haveNext = (key0 + 64) < S_;

    // ---- (a) start next K/V tile fill ---------------------------------------
    uint2 vst[8];
    if (haveNext) {
#if ATHENA_TDM
      if (wave == 0) {
        tdm_load_2d((unsigned)(uintptr_t)&ldsK[nb][0][0],
                    Kbase + (size_t)(key0 + 64) * LD,
                    (unsigned)LD, (unsigned)NT_, 64u, 64u, (unsigned)LD);
      }
#else
#pragma unroll
      for (int i = 0; i < 8; ++i) {
        int key = key0 + 64 + i * 8 + kr;
        vst[i] = *(const uint2*)(Kbase + (size_t)key * LD + eg);  // reuse below
      }
#endif
    }
    uint2 vreg[8];
    if (haveNext) {
#pragma unroll
      for (int i = 0; i < 8; ++i) {
        int key = key0 + 64 + i * 8 + kr;
        vreg[i] = *(const uint2*)(Vbase + (size_t)key * LD + eg);
      }
    }

    // ---- (b) math against buffer `cur` --------------------------------------
    f32x8 sc[4];
#pragma unroll
    for (int tt = 0; tt < 4; ++tt) {
#pragma unroll
      for (int v = 0; v < 8; ++v) sc[tt][v] = 0.0f;
      const __bf16* kc = &ldsK[cur][tt * 16 + r][0];
#pragma unroll
      for (int kk = 0; kk < 2; ++kk) {
        bf16x16 bk = make_frag(kc + kk * 32 + kb8, kc + kk * 32 + kb8 + 16);
        sc[tt] = wmma_bf16f32(aq[kk], bk, sc[tt]);
      }
    }

    // online softmax: each 16-lane half owns row v + 8*half -> xor shuffles
#pragma unroll
    for (int v = 0; v < 8; ++v) {
      float s0 = sc[0][v] * 0.125f;
      float s1 = sc[1][v] * 0.125f;
      float s2 = sc[2][v] * 0.125f;
      float s3 = sc[3][v] * 0.125f;
      float mx = fmaxf(fmaxf(s0, s1), fmaxf(s2, s3));
#pragma unroll
      for (int off = 8; off >= 1; off >>= 1)
        mx = fmaxf(mx, __shfl_xor(mx, off, 32));
      float mn = fmaxf(m[v], mx);
      float f = __expf(m[v] - mn);
      m[v] = mn;
      float p0 = __expf(s0 - mn), p1 = __expf(s1 - mn);
      float p2 = __expf(s2 - mn), p3 = __expf(s3 - mn);
      sc[0][v] = p0; sc[1][v] = p1; sc[2][v] = p2; sc[3][v] = p3;
      float rs = (p0 + p1) + (p2 + p3);
#pragma unroll
      for (int off = 8; off >= 1; off >>= 1)
        rs += __shfl_xor(rs, off, 32);
      l[v] = l[v] * f + rs;
#pragma unroll
      for (int te = 0; te < 4; ++te) o[te][v] *= f;
    }

    // P: C-layout -> A-layout via wave-private LDS staging
#pragma unroll
    for (int tt = 0; tt < 4; ++tt)
#pragma unroll
      for (int v = 0; v < 8; ++v)
        ldsP[wave][v + half * 8][tt * 16 + r] = (__bf16)sc[tt][v];

    bf16x16 ap[2];
#pragma unroll
    for (int kk = 0; kk < 2; ++kk)
      ap[kk] = make_frag(&ldsP[wave][r][kk * 32 + kb8],
                         &ldsP[wave][r][kk * 32 + kb8 + 16]);

#pragma unroll
    for (int te = 0; te < 4; ++te) {
      const __bf16* vc = &ldsVt[cur][te * 16 + r][0];
#pragma unroll
      for (int kk = 0; kk < 2; ++kk) {
        bf16x16 bv = make_frag(vc + kk * 32 + kb8, vc + kk * 32 + kb8 + 16);
        o[te] = wmma_bf16f32(ap[kk], bv, o[te]);
      }
    }

    // ---- (c) finish next tile fill ------------------------------------------
    if (haveNext) {
#if !ATHENA_TDM
#pragma unroll
      for (int i = 0; i < 8; ++i) {
        int key = i * 8 + kr;
        *(uint2*)&ldsK[nb][key][eg] = vst[i];
      }
#endif
#pragma unroll
      for (int i = 0; i < 8; ++i) {
        int key = i * 8 + kr;
        __bf16 tmp4[4];
        *(uint2*)tmp4 = vreg[i];
#pragma unroll
        for (int j = 0; j < 4; ++j) ldsVt[nb][eg + j][key] = tmp4[j];
      }
    }
    (void)vst;
#if ATHENA_TDM
    __builtin_amdgcn_s_wait_tensorcnt(0);
#endif
    __syncthreads();
    cur = nb;
  }

#pragma unroll
  for (int te = 0; te < 4; ++te) {
#pragma unroll
    for (int v = 0; v < 8; ++v) {
      int srow = q0 + v + half * 8;
      int e = te * 16 + r;
      float val = o[te][v] / l[v];
      Yg[(size_t)(b * S_ + srow) * LD + h * DH_ + e] = (__bf16)val;
    }
  }
}

// ---------------------------------------------------------------------------
// LN(in)*gamma+beta (no eps, matches reference). One block per 512-col row.
// ---------------------------------------------------------------------------
__global__ __launch_bounds__(256)
void ln_kernel(const float* __restrict__ in, const float* __restrict__ gamma,
               const float* __restrict__ beta, float* __restrict__ outF,
               __bf16* __restrict__ outB) {
  __shared__ float red[256];
  const int row = blockIdx.x;
  const int t = threadIdx.x;
  const float* pr = in + (size_t)row * D_;
  float a = pr[t];
  float b = pr[t + 256];
  red[t] = a + b;
  __syncthreads();
  for (int s2 = 128; s2 > 0; s2 >>= 1) {
    if (t < s2) red[t] += red[t + s2];
    __syncthreads();
  }
  float mean = red[0] * (1.0f / D_);
  __syncthreads();
  float da = a - mean, db = b - mean;
  red[t] = da * da + db * db;
  __syncthreads();
  for (int s2 = 128; s2 > 0; s2 >>= 1) {
    if (t < s2) red[t] += red[t + s2];
    __syncthreads();
  }
  float rstd = rsqrtf(red[0] * (1.0f / D_));
  float o1 = gamma[t] * da * rstd + beta[t];
  float o2 = gamma[t + 256] * db * rstd + beta[t + 256];
  outF[(size_t)row * D_ + t] = o1;
  outF[(size_t)row * D_ + t + 256] = o2;
  if (outB) {
    outB[(size_t)row * D_ + t] = (__bf16)o1;
    outB[(size_t)row * D_ + t + 256] = (__bf16)o2;
  }
}

// fp32 -> bf16 elementwise
__global__ void cvt_f32_bf16(const float* __restrict__ in, __bf16* __restrict__ out,
                             int n) {
  int i = blockIdx.x * 256 + threadIdx.x;
  if (i < n) out[i] = (__bf16)in[i];
}

// wq/wk/wv [H,D,DH] fp32 -> transposed [H*DH, D] bf16 (col h*64+e becomes row)
__global__ void pack_qkv_t_kernel(const float* __restrict__ in,
                                  __bf16* __restrict__ out) {
  int i = blockIdx.x * 256 + threadIdx.x;   // i = c*D + d
  if (i < D_ * D_) {
    int c = i >> 9;
    int d = i & 511;
    int h = c >> 6;
    int e = c & 63;
    out[i] = (__bf16)in[(size_t)h * D_ * DH_ + (size_t)d * DH_ + e];
  }
}

// generic [K,N] fp32 -> [N,K] bf16 transpose-convert
__global__ void pack_t_kernel(const float* __restrict__ in,
                              __bf16* __restrict__ out, int Kd, int Nd) {
  int i = blockIdx.x * 256 + threadIdx.x;   // i = n*Kd + k
  if (i < Kd * Nd) {
    int n = i / Kd;
    int k = i - n * Kd;
    out[i] = (__bf16)in[(size_t)k * Nd + n];
  }
}

extern "C" void kernel_launch(void* const* d_in, const int* in_sizes, int n_in,
                              void* d_out, int out_size, void* d_ws, size_t ws_size,
                              hipStream_t stream) {
  (void)in_sizes; (void)n_in; (void)out_size; (void)ws_size;
  const float* x   = (const float*)d_in[0];
  const float* wq  = (const float*)d_in[1];
  const float* bq  = (const float*)d_in[2];
  const float* wk  = (const float*)d_in[3];
  const float* bk  = (const float*)d_in[4];
  const float* wv  = (const float*)d_in[5];
  const float* bv  = (const float*)d_in[6];
  const float* wo  = (const float*)d_in[7];
  const float* bo  = (const float*)d_in[8];
  const float* g1  = (const float*)d_in[9];
  const float* be1 = (const float*)d_in[10];
  const float* w1  = (const float*)d_in[11];
  const float* b1  = (const float*)d_in[12];
  const float* w2  = (const float*)d_in[13];
  const float* b2  = (const float*)d_in[14];
  const float* g2  = (const float*)d_in[15];
  const float* be2 = (const float*)d_in[16];

  char* p = (char*)d_ws;
  auto carve = [&p](size_t bytes) -> void* {
    void* q = (void*)p;
    p += (bytes + 255) & ~(size_t)255;
    return q;
  };
  __bf16* xb  = (__bf16*)carve((size_t)NT_ * D_ * 2);
  __bf16* wqT = (__bf16*)carve((size_t)D_ * D_ * 2);   // [N=512][K=512]
  __bf16* wkT = (__bf16*)carve((size_t)D_ * D_ * 2);
  __bf16* wvT = (__bf16*)carve((size_t)D_ * D_ * 2);
  __bf16* woT = (__bf16*)carve((size_t)D_ * D_ * 2);
  __bf16* w1T = (__bf16*)carve((size_t)D_ * FF_ * 2);  // [2048][512]
  __bf16* w2T = (__bf16*)carve((size_t)FF_ * D_ * 2);  // [512][2048]
  __bf16* Qb  = (__bf16*)carve((size_t)NT_ * D_ * 2);
  __bf16* Kb  = (__bf16*)carve((size_t)NT_ * D_ * 2);
  __bf16* Vb  = (__bf16*)carve((size_t)NT_ * D_ * 2);
  __bf16* Yb  = (__bf16*)carve((size_t)NT_ * D_ * 2);
  float*  z1  = (float*)carve((size_t)NT_ * D_ * 4);
  float*  y1  = (float*)carve((size_t)NT_ * D_ * 4);
  __bf16* y1b = (__bf16*)carve((size_t)NT_ * D_ * 2);
  __bf16* hb  = (__bf16*)carve((size_t)NT_ * FF_ * 2);
  float*  z2  = (float*)carve((size_t)NT_ * D_ * 4);

  const dim3 blk128(128), blk256(256);

  int nx = NT_ * D_;
  cvt_f32_bf16<<<dim3((nx + 255) / 256), blk256, 0, stream>>>(x, xb, nx);
  pack_qkv_t_kernel<<<dim3((D_ * D_ + 255) / 256), blk256, 0, stream>>>(wq, wqT);
  pack_qkv_t_kernel<<<dim3((D_ * D_ + 255) / 256), blk256, 0, stream>>>(wk, wkT);
  pack_qkv_t_kernel<<<dim3((D_ * D_ + 255) / 256), blk256, 0, stream>>>(wv, wvT);
  pack_t_kernel<<<dim3((D_ * D_ + 255) / 256), blk256, 0, stream>>>(wo, woT, D_, D_);
  pack_t_kernel<<<dim3((D_ * FF_ + 255) / 256), blk256, 0, stream>>>(w1, w1T, D_, FF_);
  pack_t_kernel<<<dim3((FF_ * D_ + 255) / 256), blk256, 0, stream>>>(w2, w2T, FF_, D_);

  // QKV projections: [4096,512] x [512,512]
  gemm_bf16_kernel<<<dim3(D_ / 128, NT_ / 64), blk128, 0, stream>>>(
      xb, wqT, NT_, D_, D_, bq, nullptr, 0, nullptr, Qb);
  gemm_bf16_kernel<<<dim3(D_ / 128, NT_ / 64), blk128, 0, stream>>>(
      xb, wkT, NT_, D_, D_, bk, nullptr, 0, nullptr, Kb);
  gemm_bf16_kernel<<<dim3(D_ / 128, NT_ / 64), blk128, 0, stream>>>(
      xb, wvT, NT_, D_, D_, bv, nullptr, 0, nullptr, Vb);

  // Flash attention
  flash_attn_kernel<<<dim3(S_ / 64, B_ * H_), blk128, 0, stream>>>(Qb, Kb, Vb, Yb);

  // Out-projection + bias + residual(x) -> z1 ; LN1 -> y1 (f32 + bf16)
  gemm_bf16_kernel<<<dim3(D_ / 128, NT_ / 64), blk128, 0, stream>>>(
      Yb, woT, NT_, D_, D_, bo, x, 0, z1, nullptr);
  ln_kernel<<<dim3(NT_), blk256, 0, stream>>>(z1, g1, be1, y1, y1b);

  // FFN
  gemm_bf16_kernel<<<dim3(FF_ / 128, NT_ / 64), blk128, 0, stream>>>(
      y1b, w1T, NT_, FF_, D_, b1, nullptr, 1, nullptr, hb);
  gemm_bf16_kernel<<<dim3(D_ / 128, NT_ / 64), blk128, 0, stream>>>(
      hb, w2T, NT_, D_, FF_, b2, y1, 0, z2, nullptr);
  ln_kernel<<<dim3(NT_), blk256, 0, stream>>>(z2, g2, be2, (float*)d_out, nullptr);
}
